// GNNDrugInteractionModel_9199819948311
// MI455X (gfx1250) — matrix operand order repaired and checked by reading, not compiled, not run on hardware
//
#include <hip/hip_runtime.h>
#include <hip/hip_bf16.h>

typedef __attribute__((ext_vector_type(16))) __bf16 v16bf;
typedef __attribute__((ext_vector_type(8)))  __bf16 bf16x8;
typedef __attribute__((ext_vector_type(8)))  float  v8f;
typedef __attribute__((ext_vector_type(4)))  unsigned int u32x4;
typedef __attribute__((ext_vector_type(8)))  int i32x8;
typedef __attribute__((ext_vector_type(4)))  int i32x4;

#define BB   4096
#define NN   64
#define EE   256
#define FF   128
#define HH   64
#define FPD  2048
#define KTOT (2*HH + 2*FPD)   /* 4224 */
#define MH   256              /* MLP hidden */

// ---------- WMMA fragment helpers (layouts per CDNA5 ISA) ----------

// A-matrix 16x32 bf16 tile: lane l -> M = m0 + (l&15); half = l>>4.
__device__ __forceinline__ v16bf load_a16(const __bf16* base, int ld, int m0, int k0, int lane) {
  const int m = m0 + (lane & 15);
  const int h = lane >> 4;
  const __bf16* p = base + m * ld + k0 + 8 * h;
  bf16x8 lo = *(const bf16x8*)(p);
  bf16x8 hi = *(const bf16x8*)(p + 16);
  v16bf r;
#pragma unroll
  for (int i = 0; i < 8; ++i) { r[i] = lo[i]; r[i + 8] = hi[i]; }
  return r;
}

// B-matrix 32x16 bf16 tile read from an [N][K] (K-contiguous) layout.
__device__ __forceinline__ v16bf load_b16(const __bf16* baseT, long ld, int n0, int k0, int lane) {
  const int n = n0 + (lane & 15);
  const int h = lane >> 4;
  const __bf16* p = baseT + (long)n * ld + k0 + 8 * h;
  bf16x8 lo = *(const bf16x8*)(p);
  bf16x8 hi = *(const bf16x8*)(p + 16);
  v16bf r;
#pragma unroll
  for (int i = 0; i < 8; ++i) { r[i] = lo[i]; r[i + 8] = hi[i]; }
  return r;
}

__device__ __forceinline__ v8f wmma_bf16(v16bf a, v16bf b, v8f c) {
  return __builtin_amdgcn_wmma_f32_16x16x32_bf16(false, a, false, b, (short)0, c, false, false);
}

// C/D 16x16 f32: VGPR r -> M = m0 + r + 8*(lane>>4); N = n0 + (lane&15)
__device__ __forceinline__ void store_c(float* dst, int ld, int m0, int n0, v8f acc, int lane) {
  const int n  = n0 + (lane & 15);
  const int mb = m0 + ((lane >> 4) << 3);
#pragma unroll
  for (int r = 0; r < 8; ++r) dst[(mb + r) * ld + n] = acc[r];
}

// ---------- TDM: async DMA of a 2-D bf16 tile [256 rows x 64 cols] into LDS ----------
// Descriptor layout per CDNA5 ISA (group0/group1; groups 2/3 zero for 2-D).
__device__ __forceinline__ void tdm_load_b_tile(const __bf16* gsrc, unsigned lds_off) {
  unsigned long long ga = (unsigned long long)(uintptr_t)(const void*)gsrc;
  u32x4 g0;
  g0[0] = 1u;                                            // count=1, user descriptor
  g0[1] = lds_off;                                       // LDS byte address
  g0[2] = (unsigned)(ga & 0xffffffffu);                  // global_addr[31:0]
  g0[3] = (unsigned)((ga >> 32) & 0x1ffffffu) | (2u << 30); // global_addr[56:32] | type=2
  i32x8 g1;
  g1[0] = 0x00010000;          // workgroup_mask=0, data_size=1 (2 bytes)
  g1[1] = (int)(0x1080u << 16);// tensor_dim0 = 4224 (bits [63:48] hold low16)
  g1[2] = (int)(0x0100u << 16);// tensor_dim0 hi16 = 0; tensor_dim1 = 256 (low16 at [95:80])
  g1[3] = (int)(64u << 16);    // tensor_dim1 hi16 = 0; tile_dim0 = 64  ([127:112])
  g1[4] = 0x00000100;          // tile_dim1 = 256; tile_dim2 = 0
  g1[5] = 4224;                // tensor_dim0_stride[31:0]
  g1[6] = 0;                   // stride hi16 = 0; tensor_dim1_stride low = 0 (unused, 2-D)
  g1[7] = 0;
  i32x4 gz4 = {0, 0, 0, 0};
  i32x8 gz8 = {0, 0, 0, 0, 0, 0, 0, 0};
  __builtin_amdgcn_tensor_load_to_lds(g0, g1, gz4, gz4, gz8, 0);
}

// ---------- Kernel 0: Wm1 [4224][256] f32 -> Wm1T [256][4224] bf16 ----------
__global__ __launch_bounds__(256) void transpose_wm1(const float* __restrict__ Wm1,
                                                     __bf16* __restrict__ wm1T) {
  const int stride = gridDim.x * blockDim.x;
  for (int i = blockIdx.x * blockDim.x + threadIdx.x; i < KTOT * MH; i += stride) {
    int k = i >> 8;          // /256
    int n = i & 255;
    wm1T[(long)n * KTOT + k] = (__bf16)Wm1[i];
  }
}

// ---------- Kernel 1: per-graph GCN encoder (1 block = 1 drug graph) ----------
__global__ __launch_bounds__(256) void encoder_kernel(
    const float* __restrict__ x1, const float* __restrict__ x2,
    const int* __restrict__ edge1, const int* __restrict__ edge2,
    const float* __restrict__ W1, const float* __restrict__ b1,
    const float* __restrict__ W2, const float* __restrict__ b2,
    float* __restrict__ hbuf)
{
  __shared__ alignas(16) __bf16 xb[NN * FF];     // A for GEMM1
  __shared__ alignas(16) __bf16 w1T[HH * FF];    // B for GEMM1, [n][k]
  __shared__ alignas(16) __bf16 w2T[HH * HH];    // B for GEMM2, [n][k]
  __shared__ alignas(16) __bf16 hb[NN * HH];     // A for GEMM2
  __shared__ alignas(16) float  xw[NN * HH];     // GEMM outputs
  __shared__ alignas(16) float  agg[NN * HH];    // message accumulator
  __shared__ float deg[NN], nrm[NN], b1s[HH], b2s[HH];

  const int tid  = threadIdx.x;
  const int lane = tid & 31;
  const int wave = tid >> 5;
  const int pair = blockIdx.x >> 1;
  const int drug = blockIdx.x & 1;

  const float* x  = (drug ? x2 : x1) + (size_t)pair * NN * FF;
  const int*   ed = (drug ? edge2 : edge1) + (size_t)pair * 2 * EE;

  for (int i = tid; i < HH * FF; i += 256) {
    int n = i / FF, k = i % FF;
    w1T[i] = (__bf16)W1[k * HH + n];
  }
  for (int i = tid; i < HH * HH; i += 256) {
    int n = i / HH, k = i % HH;
    w2T[i] = (__bf16)W2[k * HH + n];
  }
  if (tid < HH) { b1s[tid] = b1[tid]; b2s[tid] = b2[tid]; }
  for (int i = tid; i < NN * FF; i += 256) xb[i] = (__bf16)x[i];
  if (tid < NN) deg[tid] = 1.0f;                 // self-loop
  __syncthreads();

  { int d = ed[EE + tid]; atomicAdd(&deg[d], 1.0f); }   // EE == blockDim
  __syncthreads();
  if (tid < NN) nrm[tid] = rsqrtf(deg[tid]);
  __syncthreads();

  // ---- GEMM1: xw = x @ W1   (M=64, N=64, K=128), 2 tiles per wave ----
  {
    const int mT  = wave >> 1;
    const int nT0 = (wave & 1) * 2;
    v8f acc0 = {}, acc1 = {};
#pragma unroll
    for (int k0 = 0; k0 < FF; k0 += 32) {
      v16bf a  = load_a16(xb, FF, mT * 16, k0, lane);
      v16bf bA = load_b16(w1T, FF, nT0 * 16, k0, lane);
      v16bf bB = load_b16(w1T, FF, (nT0 + 1) * 16, k0, lane);
      acc0 = wmma_bf16(a, bA, acc0);
      acc1 = wmma_bf16(a, bB, acc1);
    }
    store_c(xw, HH, mT * 16, nT0 * 16, acc0, lane);
    store_c(xw, HH, mT * 16, (nT0 + 1) * 16, acc1, lane);
  }
  __syncthreads();

  // ---- aggregation 1: self-loops then edge scatter (ds_add_f32) ----
  for (int i = tid; i < NN * HH; i += 256) {
    int n = i >> 6;
    agg[i] = xw[i] * nrm[n] * nrm[n];
  }
  __syncthreads();
  {
    int s = ed[tid], d = ed[EE + tid];
    float w = nrm[s] * nrm[d];
    const float* sr = &xw[s * HH];
    float* dr = &agg[d * HH];
#pragma unroll 4
    for (int c = 0; c < HH; ++c) atomicAdd(&dr[c], sr[c] * w);
  }
  __syncthreads();

  for (int i = tid; i < NN * HH; i += 256)
    hb[i] = (__bf16)fmaxf(agg[i] + b1s[i & (HH - 1)], 0.0f);
  __syncthreads();

  // ---- GEMM2: xw = h @ W2   (M=64, N=64, K=64) ----
  {
    const int mT  = wave >> 1;
    const int nT0 = (wave & 1) * 2;
    v8f acc0 = {}, acc1 = {};
#pragma unroll
    for (int k0 = 0; k0 < HH; k0 += 32) {
      v16bf a  = load_a16(hb, HH, mT * 16, k0, lane);
      v16bf bA = load_b16(w2T, HH, nT0 * 16, k0, lane);
      v16bf bB = load_b16(w2T, HH, (nT0 + 1) * 16, k0, lane);
      acc0 = wmma_bf16(a, bA, acc0);
      acc1 = wmma_bf16(a, bB, acc1);
    }
    store_c(xw, HH, mT * 16, nT0 * 16, acc0, lane);
    store_c(xw, HH, mT * 16, (nT0 + 1) * 16, acc1, lane);
  }
  __syncthreads();

  // ---- aggregation 2 ----
  for (int i = tid; i < NN * HH; i += 256) {
    int n = i >> 6;
    agg[i] = xw[i] * nrm[n] * nrm[n];
  }
  __syncthreads();
  {
    int s = ed[tid], d = ed[EE + tid];
    float w = nrm[s] * nrm[d];
    const float* sr = &xw[s * HH];
    float* dr = &agg[d * HH];
#pragma unroll 4
    for (int c = 0; c < HH; ++c) atomicAdd(&dr[c], sr[c] * w);
  }
  __syncthreads();

  if (tid < HH) {
    float s = 0.0f;
#pragma unroll 8
    for (int n = 0; n < NN; ++n) s += fmaxf(agg[n * HH + tid] + b2s[tid], 0.0f);
    hbuf[(size_t)pair * 128 + drug * 64 + tid] = s * (1.0f / NN);
  }
}

// ---------- Kernel 2: fused MLP  out = sigmoid(relu(fused@Wm1+bm1)@Wm2+bm2) ----------
// fused row = [h1(64) | h2(64) | fp1(2048) | fp2(2048)], K = 4224.
// One block = 16 rows x 256 cols. B tiles (256x64 bf16, 32KB) arrive via TDM
// (tensor_load_to_lds) into a double buffer; A tiles staged f32->bf16 by all waves.
__global__ __launch_bounds__(256) void mlp_kernel(
    const float* __restrict__ hbuf, const float* __restrict__ fp1,
    const float* __restrict__ fp2, const __bf16* __restrict__ wm1T,
    const float* __restrict__ bm1, const float* __restrict__ wm2,
    const float* __restrict__ bm2, float* __restrict__ out)
{
  __shared__ alignas(16) __bf16 aT[2][16 * 64];    // A tile double buffer
  __shared__ alignas(16) __bf16 bT[2][MH * 64];    // B tile double buffer (2 x 32KB)
  __shared__ float bm1s[MH], wm2s[MH];
  __shared__ float rowacc[16];

  const int tid  = threadIdx.x;
  const int lane = tid & 31;
  const int wave = tid >> 5;
  const int hlf  = lane >> 4;
  const int row0 = blockIdx.x * 16;

  const unsigned ldsB0 = (unsigned)(uintptr_t)(void*)&bT[0][0];
  const unsigned ldsB1 = (unsigned)(uintptr_t)(void*)&bT[1][0];

  // prologue: wave 0 kicks off the DMA of the first B tile
  if (wave == 0) tdm_load_b_tile(wm1T, ldsB0);

  bm1s[tid] = bm1[tid];
  wm2s[tid] = wm2[tid];
  if (tid < 16) rowacc[tid] = 0.0f;

  const int nT0 = 2 * wave;
  const int nT1 = 2 * wave + 1;
  const int nA  = nT0 * 16 + (lane & 15);
  const int nB  = nT1 * 16 + (lane & 15);

  v8f acc0 = {}, acc1 = {};
  const int NIT = KTOT / 64;   // 66
  for (int it = 0; it < NIT; ++it) {
    const int k0  = it * 64;
    const int cur = it & 1;

    // stage 16x64 A tile (f32 -> bf16), segment-select on k
#pragma unroll
    for (int i = tid; i < 1024; i += 256) {
      int m = i >> 6, kk = i & 63;
      int k = k0 + kk;
      int row = row0 + m;
      float v;
      if (k < 128)            v = hbuf[(size_t)row * 128 + k];
      else if (k < 128 + FPD) v = fp1[(size_t)row * FPD + (k - 128)];
      else                    v = fp2[(size_t)row * FPD + (k - 128 - FPD)];
      aT[cur][i] = (__bf16)v;
    }
    if (wave == 0) __builtin_amdgcn_s_wait_tensorcnt(0);  // bT[cur] resident
    __syncthreads();   // all waves: aT[cur]+bT[cur] ready; prior reads of bT[!cur] done
    if (wave == 0 && it + 1 < NIT)                        // DMA next B tile
      tdm_load_b_tile(wm1T + (size_t)(it + 1) * 64, cur ? ldsB0 : ldsB1);

#pragma unroll
    for (int s = 0; s < 2; ++s) {
      v16bf a  = load_a16(&aT[cur][0], 64, 0, s * 32, lane);
      v16bf b0 = load_b16(&bT[cur][0], 64, nT0 * 16, s * 32, lane);
      v16bf b1 = load_b16(&bT[cur][0], 64, nT1 * 16, s * 32, lane);
      acc0 = wmma_bf16(a, b0, acc0);
      acc1 = wmma_bf16(a, b1, acc1);
    }
  }

  // epilogue: relu(acc + bm1) dot wm2, reduce per-row via LDS atomics
  float partial[8];
#pragma unroll
  for (int r = 0; r < 8; ++r) {
    float h0 = fmaxf(acc0[r] + bm1s[nA], 0.0f);
    float h1 = fmaxf(acc1[r] + bm1s[nB], 0.0f);
    partial[r] = h0 * wm2s[nA] + h1 * wm2s[nB];
  }
  __syncthreads();
#pragma unroll
  for (int r = 0; r < 8; ++r) atomicAdd(&rowacc[r + 8 * hlf], partial[r]);
  __syncthreads();
  if (tid < 16) {
    float v = rowacc[tid] + bm2[0];
    out[row0 + tid] = 1.0f / (1.0f + __expf(-v));
  }
}

// ---------- host launcher ----------
extern "C" void kernel_launch(void* const* d_in, const int* in_sizes, int n_in,
                              void* d_out, int out_size, void* d_ws, size_t ws_size,
                              hipStream_t stream) {
  (void)in_sizes; (void)n_in; (void)out_size; (void)ws_size;
  const float* x1    = (const float*)d_in[0];
  const float* x2    = (const float*)d_in[1];
  const int*   edge1 = (const int*)d_in[2];
  const int*   edge2 = (const int*)d_in[3];
  const float* fp1   = (const float*)d_in[4];
  const float* fp2   = (const float*)d_in[5];
  const float* W1    = (const float*)d_in[6];
  const float* b1    = (const float*)d_in[7];
  const float* W2    = (const float*)d_in[8];
  const float* b2    = (const float*)d_in[9];
  const float* Wm1   = (const float*)d_in[10];
  const float* bm1   = (const float*)d_in[11];
  const float* Wm2   = (const float*)d_in[12];
  const float* bm2   = (const float*)d_in[13];
  float* out = (float*)d_out;

  float*  hbuf = (float*)d_ws;                                  // [4096][128] f32 = 2 MB
  __bf16* wm1T = (__bf16*)((char*)d_ws + (size_t)BB * 128 * 4); // [256][4224] bf16 = 2.1 MB

  transpose_wm1<<<1024, 256, 0, stream>>>(Wm1, wm1T);
  encoder_kernel<<<2 * BB, 256, 0, stream>>>(x1, x2, edge1, edge2, W1, b1, W2, b2, hbuf);
  mlp_kernel<<<BB / 16, 256, 0, stream>>>(hbuf, fp1, fp2, wm1T, bm1, Wm2, bm2, out);
}
